// SAModule_11982958755938
// MI455X (gfx1250) — matrix operand order repaired and checked by reading, not compiled, not run on hardware
//
#include <hip/hip_runtime.h>

typedef __attribute__((ext_vector_type(16))) _Float16 v16h;
typedef __attribute__((ext_vector_type(8)))  _Float16 v8h;
typedef __attribute__((ext_vector_type(8)))  float    v8f;
typedef unsigned long long u64;

#define B_   8
#define N_   4096
#define C_   64
#define M_   1024
#define K_   64
#define R2_  0.04f
#define BIG_ 1.0e10f

// ---------------------------------------------------------------------------
// WMMA helpers (CDNA5 wave32, 16x16x32 f16 -> f32)
// ---------------------------------------------------------------------------
__device__ __forceinline__ v8f wmma_f16(v16h a, v16h b, v8f c) {
    return __builtin_amdgcn_wmma_f32_16x16x32_f16(false, a, false, b,
                                                  (short)0, c, false, false);
}

// A fragment (16x32, M x K), source row-major [rows][strideHalves] in LDS.
// ISA layout: lanes 0-15 hold K={0..7,16..23}, lanes 16-31 hold K={8..15,24..31}.
__device__ __forceinline__ v16h frag_a_lds(const _Float16* base, int row0,
                                           int kOff, int stride) {
    const int lane = threadIdx.x & 31;
    const int r = lane & 15, hi = lane >> 4;
    const _Float16* p = base + (row0 + r) * stride + kOff + hi * 8;
    v8h lo  = *(const v8h*)(p);
    v8h hi8 = *(const v8h*)(p + 16);
    v16h o;
#pragma unroll
    for (int i = 0; i < 8; ++i) { o[i] = lo[i]; o[i + 8] = hi8[i]; }
    return o;
}

// B fragment (32x16, K x N), source W^T row-major [Nout][strideHalves] in global.
// ISA layout: lanes 0-15 hold K={0..7,16..23} of column n, lanes 16-31 hold K={8..15,24..31}.
__device__ __forceinline__ v16h frag_b_glb(const _Float16* baseT, int n0,
                                           int kOff, int stride) {
    const int lane = threadIdx.x & 31;
    const int n = lane & 15, hi = lane >> 4;
    const _Float16* p = baseT + (size_t)(n0 + n) * stride + kOff + hi * 8;
    v8h lo  = *(const v8h*)(p);
    v8h hi8 = *(const v8h*)(p + 16);
    v16h o;
#pragma unroll
    for (int i = 0; i < 8; ++i) { o[i] = lo[i]; o[i + 8] = hi8[i]; }
    return o;
}

// D tile (16x16): lane&15 = column, vgpr j = row j + 8*(lane>=16). Bias+ReLU,
// convert to f16, store row-major into LDS.
__device__ __forceinline__ void store_relu_f16(_Float16* base, int row0, int col0,
                                               int stride, v8f acc, float bias) {
    const int lane = threadIdx.x & 31;
    const int c = lane & 15, hi = lane >> 4;
#pragma unroll
    for (int j = 0; j < 8; ++j) {
        float v = acc[j] + bias;
        v = v > 0.f ? v : 0.f;
        base[(row0 + hi * 8 + j) * stride + col0 + c] = (_Float16)v;
    }
}

// ---------------------------------------------------------------------------
// Kernel 0: transpose + f16-convert weights into workspace
//   W1 [67][128] -> W1T f16 [128][96] (K padded 67->96 with zeros)
//   W2 [128][128] -> W2T f16 [128][128]
//   W3 [128][256] -> W3T f16 [256][128]
// ---------------------------------------------------------------------------
__global__ void prep_kernel(const float* __restrict__ W1, const float* __restrict__ W2,
                            const float* __restrict__ W3, _Float16* __restrict__ W1T,
                            _Float16* __restrict__ W2T, _Float16* __restrict__ W3T) {
    int t = blockIdx.x * blockDim.x + threadIdx.x;
    if (t < 128 * 96) {
        int n = t / 96, k = t % 96;
        W1T[t] = (_Float16)(k < 67 ? W1[k * 128 + n] : 0.f);
    } else if (t < 128 * 96 + 128 * 128) {
        int t2 = t - 128 * 96;
        int n = t2 / 128, k = t2 % 128;
        W2T[t2] = (_Float16)W2[k * 128 + n];
    } else if (t < 128 * 96 + 128 * 128 + 256 * 128) {
        int t3 = t - 128 * 96 - 128 * 128;
        int n = t3 / 128, k = t3 % 128;
        W3T[t3] = (_Float16)W3[k * 256 + n];
    }
}

// ---------------------------------------------------------------------------
// Kernel 1: farthest point sampling, one block per graph (exact f32).
// Also emits pos_out / batch_out / seed_out and centroid positions.
// ---------------------------------------------------------------------------
__global__ void __launch_bounds__(256) fps_kernel(const float* __restrict__ pos,
                                                  int* __restrict__ idxLocal,
                                                  float* __restrict__ ctr,
                                                  float* __restrict__ posOut,
                                                  float* __restrict__ batchOut,
                                                  float* __restrict__ seedOut) {
    __shared__ u64 sBest;
    __shared__ float sPos[3];
    const int b = blockIdx.x, t = threadIdx.x;
    const int base = t * 16;
    const float* pg = pos + (size_t)b * N_ * 3;

    float px[16], py[16], pz[16], mind[16];
#pragma unroll
    for (int i = 0; i < 16; ++i) {
        px[i] = pg[(base + i) * 3 + 0];
        py[i] = pg[(base + i) * 3 + 1];
        pz[i] = pg[(base + i) * 3 + 2];
        mind[i] = BIG_;
    }
    if (t == 0) {
        sBest = 0ull;
        sPos[0] = pg[0]; sPos[1] = pg[1]; sPos[2] = pg[2];
        int o = b * M_;
        idxLocal[o] = 0;
        ctr[(size_t)o * 3 + 0] = pg[0]; ctr[(size_t)o * 3 + 1] = pg[1]; ctr[(size_t)o * 3 + 2] = pg[2];
        posOut[(size_t)o * 3 + 0] = pg[0]; posOut[(size_t)o * 3 + 1] = pg[1]; posOut[(size_t)o * 3 + 2] = pg[2];
        batchOut[o] = (float)b;
        seedOut[o]  = (float)(b * N_);
    }
    __syncthreads();

    for (int s = 1; s < M_; ++s) {
        const float lx = sPos[0], ly = sPos[1], lz = sPos[2];
        u64 best = 0ull;
#pragma unroll
        for (int i = 0; i < 16; ++i) {
            float dx = px[i] - lx, dy = py[i] - ly, dz = pz[i] - lz;
            float d = dx * dx + dy * dy + dz * dz;
            mind[i] = fminf(mind[i], d);
            u64 key = ((u64)__float_as_uint(mind[i]) << 32) |
                      (u64)(0xFFFFFFFFu - (unsigned)(base + i));
            best = key > best ? key : best;
        }
#pragma unroll
        for (int off = 16; off > 0; off >>= 1) {
            u64 o = __shfl_xor(best, off);
            best = o > best ? o : best;
        }
        if ((t & 31) == 0) atomicMax(&sBest, best);
        __syncthreads();
        const u64 k = sBest;
        const int win = (int)(0xFFFFFFFFu - (unsigned)(k & 0xFFFFFFFFull));
#pragma unroll
        for (int i = 0; i < 16; ++i) {
            if (base + i == win) { sPos[0] = px[i]; sPos[1] = py[i]; sPos[2] = pz[i]; }
        }
        __syncthreads();
        if (t == 0) {
            sBest = 0ull;
            int o = b * M_ + s;
            idxLocal[o] = win;
            ctr[(size_t)o * 3 + 0] = sPos[0]; ctr[(size_t)o * 3 + 1] = sPos[1]; ctr[(size_t)o * 3 + 2] = sPos[2];
            posOut[(size_t)o * 3 + 0] = sPos[0]; posOut[(size_t)o * 3 + 1] = sPos[1]; posOut[(size_t)o * 3 + 2] = sPos[2];
            batchOut[o] = (float)b;
            seedOut[o]  = (float)(b * N_ + win);
        }
        __syncthreads();
    }
}

// ---------------------------------------------------------------------------
// Kernel 2: radius top-K (K nearest within R), exact f32, 8 centroids/block.
// nbr[c*K + r] = point index, or -1 when invalid.
// ---------------------------------------------------------------------------
__global__ void __launch_bounds__(256) radius_kernel(const float* __restrict__ pos,
                                                     const float* __restrict__ ctr,
                                                     int* __restrict__ nbr) {
    __shared__ float sD[N_];
    __shared__ u64 sKey;
    const int t = threadIdx.x;
    const int cbase = blockIdx.x * 8;

    for (int ci = 0; ci < 8; ++ci) {
        const int cg = cbase + ci;
        const int b = cg >> 10;
        const float cx = ctr[cg * 3 + 0], cy = ctr[cg * 3 + 1], cz = ctr[cg * 3 + 2];
        const float* pg = pos + (size_t)b * N_ * 3;
        for (int i = t; i < N_; i += 256) {
            float dx = pg[i * 3 + 0] - cx, dy = pg[i * 3 + 1] - cy, dz = pg[i * 3 + 2] - cz;
            float d = dx * dx + dy * dy + dz * dz;
            sD[i] = d < R2_ ? d : BIG_;
        }
        if (t == 0) sKey = ~0ull;
        __syncthreads();

        int done = 0;
        for (int r = 0; r < K_; ++r) {
            if (!done) {
                u64 best = ~0ull;
                for (int i = t; i < N_; i += 256) {
                    u64 key = ((u64)__float_as_uint(sD[i]) << 32) | (unsigned)i;
                    best = key < best ? key : best;
                }
#pragma unroll
                for (int off = 16; off > 0; off >>= 1) {
                    u64 o = __shfl_xor(best, off);
                    best = o < best ? o : best;
                }
                if ((t & 31) == 0) atomicMin(&sKey, best);
            }
            __syncthreads();
            const u64 k = sKey;
            __syncthreads();
            const float v = __uint_as_float((unsigned)(k >> 32));
            const int widx = (int)(unsigned)(k & 0xFFFFFFFFull);
            if (v < R2_ && !done) {
                if (t == 0) {
                    nbr[(size_t)cg * K_ + r] = widx;
                    sD[widx] = 2.f * BIG_;
                    sKey = ~0ull;
                }
            } else {
                if (t == 0) { nbr[(size_t)cg * K_ + r] = -1; sKey = ~0ull; }
                done = 1;
            }
            __syncthreads();
        }
    }
}

// ---------------------------------------------------------------------------
// Kernel 3: fused 3-layer PointConv MLP + max aggregation.
// One block (4 waves / 128 threads) per centroid; 64 neighbor rows in LDS.
// ---------------------------------------------------------------------------
__global__ void __launch_bounds__(128) mlp_kernel(
    const float* __restrict__ x, const float* __restrict__ pos,
    const float* __restrict__ ctr, const int* __restrict__ nbr,
    const _Float16* __restrict__ W1T, const _Float16* __restrict__ W2T,
    const _Float16* __restrict__ W3T, const float* __restrict__ b1,
    const float* __restrict__ b2, const float* __restrict__ b3,
    float* __restrict__ xout) {
    __shared__ _Float16 sMsg[64 * 104];   // rows=neighbor slots, cols 0..66 data, pad->96 zeros
    __shared__ _Float16 sH1[64 * 136];
    __shared__ _Float16 sH2[64 * 136];
    __shared__ int   sNbr[64];
    __shared__ float sMax[256];

    const int cg = blockIdx.x;
    const int b = cg >> 10;
    const int t = threadIdx.x;
    const int lane = t & 31, w = t >> 5;
    const int l15 = lane & 15, hi = lane >> 4;
    const int row0 = w * 16;

    __builtin_prefetch(W1T, 0, 2);
    __builtin_prefetch(W2T, 0, 2);
    __builtin_prefetch(W3T, 0, 2);

    const float cx = ctr[cg * 3 + 0], cy = ctr[cg * 3 + 1], cz = ctr[cg * 3 + 2];

    // ---- stage message rows into LDS (f16) ----
    {
        const int row = t & 63, part = t >> 6;
        const int j = nbr[(size_t)cg * K_ + row];
        const int jj = j < 0 ? 0 : j;
        if (part == 0) {
            sNbr[row] = j;
            const float4* xr = (const float4*)(x + ((size_t)b * N_ + jj) * C_);
#pragma unroll
            for (int c4 = 0; c4 < 16; ++c4) {
                float4 v = xr[c4];
                if (j < 0) { v.x = v.y = v.z = v.w = 0.f; }
                sMsg[row * 104 + c4 * 4 + 0] = (_Float16)v.x;
                sMsg[row * 104 + c4 * 4 + 1] = (_Float16)v.y;
                sMsg[row * 104 + c4 * 4 + 2] = (_Float16)v.z;
                sMsg[row * 104 + c4 * 4 + 3] = (_Float16)v.w;
            }
        } else {
            const float* pr = pos + ((size_t)b * N_ + jj) * 3;
            sMsg[row * 104 + 64] = (j >= 0) ? (_Float16)(pr[0] - cx) : (_Float16)0.f;
            sMsg[row * 104 + 65] = (j >= 0) ? (_Float16)(pr[1] - cy) : (_Float16)0.f;
            sMsg[row * 104 + 66] = (j >= 0) ? (_Float16)(pr[2] - cz) : (_Float16)0.f;
            for (int c = 67; c < 96; ++c) sMsg[row * 104 + c] = (_Float16)0.f;
        }
        sMax[t] = 0.f;
        sMax[t + 128] = 0.f;
    }
    __syncthreads();

    // ---- layer 1: [64 x 96(f16)] @ W1T -> h1 [64 x 128] ----
    {
        v16h a0 = frag_a_lds(sMsg, row0, 0, 104);
        v16h a1 = frag_a_lds(sMsg, row0, 32, 104);
        v16h a2 = frag_a_lds(sMsg, row0, 64, 104);
#pragma unroll
        for (int n = 0; n < 8; ++n) {
            v8f acc = {0, 0, 0, 0, 0, 0, 0, 0};
            acc = wmma_f16(a0, frag_b_glb(W1T, n * 16, 0, 96), acc);
            acc = wmma_f16(a1, frag_b_glb(W1T, n * 16, 32, 96), acc);
            acc = wmma_f16(a2, frag_b_glb(W1T, n * 16, 64, 96), acc);
            store_relu_f16(sH1, row0, n * 16, 136, acc, b1[n * 16 + l15]);
        }
    }
    __syncthreads();

    // ---- layer 2: h1 @ W2T -> h2 [64 x 128] ----
    {
        v16h a0 = frag_a_lds(sH1, row0, 0, 136);
        v16h a1 = frag_a_lds(sH1, row0, 32, 136);
        v16h a2 = frag_a_lds(sH1, row0, 64, 136);
        v16h a3 = frag_a_lds(sH1, row0, 96, 136);
#pragma unroll
        for (int n = 0; n < 8; ++n) {
            v8f acc = {0, 0, 0, 0, 0, 0, 0, 0};
            acc = wmma_f16(a0, frag_b_glb(W2T, n * 16, 0, 128), acc);
            acc = wmma_f16(a1, frag_b_glb(W2T, n * 16, 32, 128), acc);
            acc = wmma_f16(a2, frag_b_glb(W2T, n * 16, 64, 128), acc);
            acc = wmma_f16(a3, frag_b_glb(W2T, n * 16, 96, 128), acc);
            store_relu_f16(sH2, row0, n * 16, 136, acc, b2[n * 16 + l15]);
        }
    }
    __syncthreads();

    // ---- layer 3: h2 @ W3T -> [64 x 256], relu, mask, max over 64 rows ----
    {
        v16h a0 = frag_a_lds(sH2, row0, 0, 136);
        v16h a1 = frag_a_lds(sH2, row0, 32, 136);
        v16h a2 = frag_a_lds(sH2, row0, 64, 136);
        v16h a3 = frag_a_lds(sH2, row0, 96, 136);
        unsigned vmask = 0;
#pragma unroll
        for (int j = 0; j < 8; ++j)
            if (sNbr[row0 + hi * 8 + j] >= 0) vmask |= 1u << j;
#pragma unroll
        for (int n = 0; n < 16; ++n) {
            v8f acc = {0, 0, 0, 0, 0, 0, 0, 0};
            acc = wmma_f16(a0, frag_b_glb(W3T, n * 16, 0, 128), acc);
            acc = wmma_f16(a1, frag_b_glb(W3T, n * 16, 32, 128), acc);
            acc = wmma_f16(a2, frag_b_glb(W3T, n * 16, 64, 128), acc);
            acc = wmma_f16(a3, frag_b_glb(W3T, n * 16, 96, 128), acc);
            const float bias = b3[n * 16 + l15];
            float m = 0.f;
#pragma unroll
            for (int j = 0; j < 8; ++j) {
                float v = acc[j] + bias;
                v = v > 0.f ? v : 0.f;                  // relu
                if ((vmask >> j) & 1) m = fmaxf(m, v);  // invalid rows -> 0
            }
            m = fmaxf(m, __shfl_xor(m, 16));
            if (lane < 16)  // values >= 0, so int compare == float compare
                atomicMax((int*)&sMax[n * 16 + l15], __float_as_int(m));
        }
    }
    __syncthreads();

    for (int c = t; c < 256; c += 128) xout[(size_t)cg * 256 + c] = sMax[c];
}

// ---------------------------------------------------------------------------
extern "C" void kernel_launch(void* const* d_in, const int* in_sizes, int n_in,
                              void* d_out, int out_size, void* d_ws, size_t ws_size,
                              hipStream_t stream) {
    (void)in_sizes; (void)n_in; (void)out_size; (void)ws_size;
    const float* x   = (const float*)d_in[0];
    const float* pos = (const float*)d_in[1];
    const float* W1  = (const float*)d_in[4];
    const float* b1  = (const float*)d_in[5];
    const float* W2  = (const float*)d_in[6];
    const float* b2  = (const float*)d_in[7];
    const float* W3  = (const float*)d_in[8];
    const float* b3  = (const float*)d_in[9];

    float* out      = (float*)d_out;
    float* xout     = out;                                   // [8192, 256]
    float* posOut   = out + (size_t)B_ * M_ * 256;           // [8192, 3]
    float* batchOut = posOut + (size_t)B_ * M_ * 3;          // [8192]
    float* seedOut  = batchOut + (size_t)B_ * M_;            // [8192]

    char* ws = (char*)d_ws;
    int*      idxLocal = (int*)(ws);                         //  32 KB
    float*    ctr      = (float*)(ws + 32768);               //  96 KB
    int*      nbr      = (int*)(ws + 131072);                //   2 MB
    _Float16* W1T      = (_Float16*)(ws + 2228224);          //  24 KB
    _Float16* W2T      = (_Float16*)(ws + 2252800);          //  32 KB
    _Float16* W3T      = (_Float16*)(ws + 2285568);          //  64 KB

    prep_kernel<<<(128 * 96 + 128 * 128 + 256 * 128 + 255) / 256, 256, 0, stream>>>(
        W1, W2, W3, W1T, W2T, W3T);
    fps_kernel<<<B_, 256, 0, stream>>>(pos, idxLocal, ctr, posOut, batchOut, seedOut);
    radius_kernel<<<(B_ * M_) / 8, 256, 0, stream>>>(pos, ctr, nbr);
    mlp_kernel<<<B_ * M_, 128, 0, stream>>>(x, pos, ctr, nbr, W1T, W2T, W3T,
                                            b1, b2, b3, xout);
}
